// MultiHeadedAttention_62603443306891
// MI455X (gfx1250) — matrix operand order repaired and checked by reading, not compiled, not run on hardware
//
#include <hip/hip_runtime.h>
#include <hip/hip_bf16.h>

// CDNA5 / gfx1250, wave32. All matmuls via v_wmma_f32_16x16x32_bf16.
// Each wave computes a 32x32 output tile (2x2 WMMA tiles). Long K loops are
// manually double-buffered (compute set A while loading set B and vice
// versa, no register copies) and kept rolled so only one prefetch set is in
// flight. The short-K qkv kernel uses a simple rolled loop to avoid the
// compiler flattening a trip-count-1 pipeline and spilling.

typedef __attribute__((ext_vector_type(16))) __bf16 v16bf;
typedef __attribute__((ext_vector_type(8)))  float  v8f;

__device__ __forceinline__ v8f wmma_bf16(v16bf a, v16bf b, v8f c) {
  return __builtin_amdgcn_wmma_f32_16x16x32_bf16(false, a, false, b, (short)0, c,
                                                 false, false);
}

union frag_u { v16bf v; uint4 u[2]; };

// A fragment (16x32 bf16, M x K): lane holds row M = lane&15.
// Lanes 0-15: K = {0..7, 16..23}; lanes 16-31: K = {8..15, 24..31}.
__device__ __forceinline__ v16bf load_a_frag(const __bf16* row, int k0, int kb) {
  frag_u t;
  t.u[0] = *(const uint4*)(row + k0 + kb);
  t.u[1] = *(const uint4*)(row + k0 + 16 + kb);
  return t.v;
}

// B fragment (32x16 bf16, K x N) where B[k][n] = M2[n][k], M2 row-major.
// Lane holds column N = lane&15; lanes 0-15: K = {0..7, 16..23}; lanes 16-31: K = {8..15, 24..31}.
__device__ __forceinline__ v16bf load_bt_frag(const __bf16* row, int k0, int hi) {
  frag_u t;
  const __bf16* p = row + k0 + hi * 8;
  t.u[0] = *(const uint4*)(p);
  t.u[1] = *(const uint4*)(p + 16);
  return t.v;
}

// Double-buffered 2x2 GEMM over K (K multiple of 64, K >= 192 so the rolled
// loop has trip count >= 2). No register copies.
template <int K>
__device__ __forceinline__ void gemm_pipe(const __bf16* arow0, const __bf16* arow1,
                                          const __bf16* brow0, const __bf16* brow1,
                                          int kb, int hi, v8f (&acc)[2][2]) {
  v16bf a0 = load_a_frag(arow0, 0, kb), a1 = load_a_frag(arow1, 0, kb);
  v16bf b0 = load_bt_frag(brow0, 0, hi), b1 = load_bt_frag(brow1, 0, hi);
#pragma unroll 1
  for (int k0 = 0; k0 + 64 < K; k0 += 64) {
    v16bf a0n = load_a_frag(arow0, k0 + 32, kb);
    v16bf a1n = load_a_frag(arow1, k0 + 32, kb);
    v16bf b0n = load_bt_frag(brow0, k0 + 32, hi);
    v16bf b1n = load_bt_frag(brow1, k0 + 32, hi);
    acc[0][0] = wmma_bf16(a0, b0, acc[0][0]);
    acc[0][1] = wmma_bf16(a0, b1, acc[0][1]);
    acc[1][0] = wmma_bf16(a1, b0, acc[1][0]);
    acc[1][1] = wmma_bf16(a1, b1, acc[1][1]);
    a0 = load_a_frag(arow0, k0 + 64, kb);
    a1 = load_a_frag(arow1, k0 + 64, kb);
    b0 = load_bt_frag(brow0, k0 + 64, hi);
    b1 = load_bt_frag(brow1, k0 + 64, hi);
    acc[0][0] = wmma_bf16(a0n, b0n, acc[0][0]);
    acc[0][1] = wmma_bf16(a0n, b1n, acc[0][1]);
    acc[1][0] = wmma_bf16(a1n, b0n, acc[1][0]);
    acc[1][1] = wmma_bf16(a1n, b1n, acc[1][1]);
  }
  // epilogue: set A holds chunk K-64; load K-32 and finish both.
  v16bf a0n = load_a_frag(arow0, K - 32, kb);
  v16bf a1n = load_a_frag(arow1, K - 32, kb);
  v16bf b0n = load_bt_frag(brow0, K - 32, hi);
  v16bf b1n = load_bt_frag(brow1, K - 32, hi);
  acc[0][0] = wmma_bf16(a0, b0, acc[0][0]);
  acc[0][1] = wmma_bf16(a0, b1, acc[0][1]);
  acc[1][0] = wmma_bf16(a1, b0, acc[1][0]);
  acc[1][1] = wmma_bf16(a1, b1, acc[1][1]);
  acc[0][0] = wmma_bf16(a0n, b0n, acc[0][0]);
  acc[0][1] = wmma_bf16(a0n, b1n, acc[0][1]);
  acc[1][0] = wmma_bf16(a1n, b0n, acc[1][0]);
  acc[1][1] = wmma_bf16(a1n, b1n, acc[1][1]);
}

// ---------------- K0: x [B,C,H,W] f32 -> xT [B*HW, C] bf16 (LDS transpose) --
__global__ void k_convert_x(const float* __restrict__ x, __bf16* __restrict__ xT) {
  __shared__ __bf16 tile[128][136];
  size_t n0 = (size_t)blockIdx.x * 128;          // global pixel base (b*65536+pix)
  int b = (int)(n0 >> 16);
  int pix0 = (int)(n0 & 65535);
  const float* xb = x + (size_t)b * 128 * 65536;
  for (int i = threadIdx.x; i < 128 * 128; i += 256) {
    int c = i >> 7, pp = i & 127;                // coalesced reads along pixels
    tile[c][pp] = (__bf16)xb[(size_t)c * 65536 + pix0 + pp];
  }
  __syncthreads();
  for (int i = threadIdx.x; i < 128 * 128; i += 256) {
    int pp = i >> 7, c = i & 127;                // contiguous bf16 writes
    xT[(n0 + pp) * 128 + c] = tile[c][pp];
  }
}

// ---------------- K0b: weights f32 -> bf16 (wo reordered to [o][tap][c]) ----
__global__ void k_convert_w(const float* __restrict__ wq, const float* __restrict__ wk,
                            const float* __restrict__ wv, const float* __restrict__ wo,
                            __bf16* __restrict__ wbf, __bf16* __restrict__ wobf) {
  int tid = blockIdx.x * blockDim.x + threadIdx.x;
  int stride = gridDim.x * blockDim.x;
  const int total1 = 3 * 128 * 128;
  for (int i = tid; i < total1; i += stride) {
    int m = i / 16384, r = i % 16384;
    const float* w = (m == 0) ? wq : (m == 1) ? wk : wv;
    wbf[i] = (__bf16)w[r];
  }
  const int total2 = 128 * 1152;                 // [o][(ky*3+kx)*128 + c]
  for (int i = tid; i < total2; i += stride) {
    int o = i / 1152, rem = i % 1152;
    int tap = rem / 128, c = rem % 128;
    int ky = tap / 3, kx = tap % 3;
    wobf[i] = (__bf16)wo[(((size_t)o * 128 + c) * 3 + ky) * 3 + kx];
  }
}

// ---------------- K1: QKV projection GEMM, scatter into windowed layout -----
// q,k: [bh][s(1024)][dd(2048)] bf16 ; vT: [bh][dd(2048)][s(1024)] bf16
__global__ void k_qkv(const __bf16* __restrict__ xT, const __bf16* __restrict__ wbf,
                      const float* __restrict__ bq, const float* __restrict__ bk,
                      const float* __restrict__ bv,
                      __bf16* __restrict__ q, __bf16* __restrict__ kq,
                      __bf16* __restrict__ vT) {
  int wave = threadIdx.x >> 5, lane = threadIdx.x & 31;
  int n = lane & 15, hi = lane >> 4, kb = hi * 8;
  int m = blockIdx.y;                            // 0=q 1=k 2=v
  int o0 = wave * 32;                            // 32 output channels per wave
  size_t ng0 = (size_t)blockIdx.x * 32;          // 32 consecutive pixels
  const __bf16* wm = wbf + (size_t)m * 16384;
  const __bf16* arow0 = wm + (size_t)(o0 + n) * 128;
  const __bf16* arow1 = wm + (size_t)(o0 + 16 + n) * 128;
  const __bf16* brow0 = xT + (ng0 + n) * 128;
  const __bf16* brow1 = xT + (ng0 + 16 + n) * 128;

  // Simple rolled loop (K=128 is too short to pipeline without flattening).
  v8f acc[2][2] = {};
#pragma unroll 1
  for (int k0 = 0; k0 < 128; k0 += 32) {
    v16bf a0 = load_a_frag(arow0, k0, kb);
    v16bf a1 = load_a_frag(arow1, k0, kb);
    v16bf b0 = load_bt_frag(brow0, k0, hi);
    v16bf b1 = load_bt_frag(brow1, k0, hi);
    acc[0][0] = wmma_bf16(a0, b0, acc[0][0]);
    acc[0][1] = wmma_bf16(a0, b1, acc[0][1]);
    acc[1][0] = wmma_bf16(a1, b0, acc[1][0]);
    acc[1][1] = wmma_bf16(a1, b1, acc[1][1]);
  }

  const float* bias = (m == 0) ? bq : (m == 1) ? bk : bv;
  int b = (int)(ng0 >> 16);
  int pixbase = (int)(ng0 & 65535);
#pragma unroll
  for (int chh = 0; chh < 2; ++chh) {
    int pix = pixbase + chh * 16 + n;
    int h = pix >> 8, w = pix & 255;
    int s = (h >> 3) * 32 + (w >> 3);
    int loc = (h & 7) * 8 + (w & 7);
#pragma unroll
    for (int rh = 0; rh < 2; ++rh) {
#pragma unroll
      for (int r = 0; r < 8; ++r) {
        int o = o0 + rh * 16 + hi * 8 + r;
        float val = acc[rh][chh][r] + bias[o];
        int head = o >> 5;
        int dd = (o & 31) * 64 + loc;
        size_t bh = (size_t)(b * 4 + head);
        if (m == 0)      q [(bh * 1024 + s) * 2048 + dd] = (__bf16)val;
        else if (m == 1) kq[(bh * 1024 + s) * 2048 + dd] = (__bf16)val;
        else             vT[(bh * 2048 + dd) * 1024 + s] = (__bf16)val;
      }
    }
  }
}

// ---------------- K2: S = (Q K^T) * 1/sqrt(d) -------------------------------
__global__ void k_scores(const __bf16* __restrict__ q, const __bf16* __restrict__ kk,
                         float* __restrict__ S) {
  int wave = threadIdx.x >> 5, lane = threadIdx.x & 31;
  int n = lane & 15, hi = lane >> 4, kb = hi * 8;
  int qt2 = blockIdx.x;                          // 0..31 (32 query rows)
  int kt2 = blockIdx.y * 4 + wave;               // 0..31 (32 key rows)
  int bh = blockIdx.z;                           // 0..7
  const __bf16* qb = q  + (size_t)bh * 1024 * 2048;
  const __bf16* kbb = kk + (size_t)bh * 1024 * 2048;
  const __bf16* arow0 = qb + (size_t)(qt2 * 32 + n) * 2048;
  const __bf16* arow1 = qb + (size_t)(qt2 * 32 + 16 + n) * 2048;
  const __bf16* brow0 = kbb + (size_t)(kt2 * 32 + n) * 2048;
  const __bf16* brow1 = kbb + (size_t)(kt2 * 32 + 16 + n) * 2048;

  v8f acc[2][2] = {};
  gemm_pipe<2048>(arow0, arow1, brow0, brow1, kb, hi, acc);

  const float scale = 0.022097086912079608f;     // 1/sqrt(2048)
  float* Sb = S + (size_t)bh * 1024 * 1024;
#pragma unroll
  for (int rh = 0; rh < 2; ++rh)
#pragma unroll
    for (int chh = 0; chh < 2; ++chh)
#pragma unroll
      for (int r = 0; r < 8; ++r)
        Sb[(size_t)(qt2 * 32 + rh * 16 + hi * 8 + r) * 1024 +
           kt2 * 32 + chh * 16 + n] = acc[rh][chh][r] * scale;
}

// ---------------- K3: row softmax, write P as bf16 --------------------------
__global__ void k_softmax(const float* __restrict__ S, __bf16* __restrict__ P) {
  __shared__ float red[256];
  size_t row = blockIdx.x;
  const float* src = S + row * 1024;
  __bf16* dst = P + row * 1024;
  int t = threadIdx.x;
  float4 v = ((const float4*)src)[t];
  red[t] = fmaxf(fmaxf(v.x, v.y), fmaxf(v.z, v.w));
  __syncthreads();
  for (int s = 128; s > 0; s >>= 1) {
    if (t < s) red[t] = fmaxf(red[t], red[t + s]);
    __syncthreads();
  }
  float mx = red[0];
  __syncthreads();
  float e0 = __expf(v.x - mx), e1 = __expf(v.y - mx);
  float e2 = __expf(v.z - mx), e3 = __expf(v.w - mx);
  red[t] = e0 + e1 + e2 + e3;
  __syncthreads();
  for (int s = 128; s > 0; s >>= 1) {
    if (t < s) red[t] += red[t + s];
    __syncthreads();
  }
  float inv = 1.0f / red[0];
  dst[t * 4 + 0] = (__bf16)(e0 * inv);
  dst[t * 4 + 1] = (__bf16)(e1 * inv);
  dst[t * 4 + 2] = (__bf16)(e2 * inv);
  dst[t * 4 + 3] = (__bf16)(e3 * inv);
}

// ---------------- K4: Y = P V, scatter to pixel-major y [b][hh][ww][c] ------
__global__ void k_pv(const __bf16* __restrict__ P, const __bf16* __restrict__ vT,
                     __bf16* __restrict__ y) {
  int wave = threadIdx.x >> 5, lane = threadIdx.x & 31;
  int n = lane & 15, hi = lane >> 4, kb = hi * 8;
  int qt2 = blockIdx.x;                          // 0..31
  int ddt2 = blockIdx.y * 4 + wave;              // 0..63 (32 dd cols)
  int bh = blockIdx.z;                           // 0..7
  int dd0 = ddt2 * 32;
  const __bf16* Pb = P + (size_t)bh * 1024 * 1024;
  const __bf16* vb = vT + (size_t)bh * 2048 * 1024;
  const __bf16* arow0 = Pb + (size_t)(qt2 * 32 + n) * 1024;
  const __bf16* arow1 = Pb + (size_t)(qt2 * 32 + 16 + n) * 1024;
  const __bf16* brow0 = vb + (size_t)(dd0 + n) * 1024;
  const __bf16* brow1 = vb + (size_t)(dd0 + 16 + n) * 1024;

  v8f acc[2][2] = {};
  gemm_pipe<1024>(arow0, arow1, brow0, brow1, kb, hi, acc);

  int b = bh >> 2, head = bh & 3;
#pragma unroll
  for (int chh = 0; chh < 2; ++chh) {
    int dd = dd0 + chh * 16 + n;
    int cin = dd >> 6, ph = (dd >> 3) & 7, pw = dd & 7;
    int ch = head * 32 + cin;
#pragma unroll
    for (int rh = 0; rh < 2; ++rh) {
#pragma unroll
      for (int r = 0; r < 8; ++r) {
        int s = qt2 * 32 + rh * 16 + hi * 8 + r;
        int hh = (s >> 5) * 8 + ph, ww = (s & 31) * 8 + pw;
        y[((size_t)b * 65536 + hh * 256 + ww) * 128 + ch] = (__bf16)acc[rh][chh][r];
      }
    }
  }
}

// ---------------- K5: 3x3 reflect conv (im2col WMMA) + bias + LeakyReLU -----
__device__ __forceinline__ const __bf16* conv_src(const __bf16* yb, int hh, int ww,
                                                  int tap) {
  int ky = tap / 3 - 1, kx = tap % 3 - 1;
  int sh = hh + ky; sh = sh < 0 ? -sh : (sh > 255 ? 510 - sh : sh);     // reflect
  int sw = ww + kx; sw = sw < 0 ? -sw : (sw > 255 ? 510 - sw : sw);
  return yb + (size_t)(sh * 256 + sw) * 128;
}

__global__ void k_conv(const __bf16* __restrict__ y, const __bf16* __restrict__ wobf,
                       const float* __restrict__ bo, float* __restrict__ out) {
  int wave = threadIdx.x >> 5, lane = threadIdx.x & 31;
  int n = lane & 15, hi = lane >> 4, kb = hi * 8;
  int pt = blockIdx.x;                           // 2048 pixel tiles per batch
  int hh = pt >> 3, w0 = (pt & 7) * 32;
  int o0 = wave * 32;                            // 32 output channels per wave
  int b = blockIdx.z;
  int ww0 = w0 + n, ww1 = w0 + 16 + n;
  const __bf16* arow0 = wobf + (size_t)(o0 + n) * 1152;
  const __bf16* arow1 = wobf + (size_t)(o0 + 16 + n) * 1152;
  const __bf16* yb = y + (size_t)b * 65536 * 128;

  v8f acc[2][2] = {};
  // 36 k-chunks (9 taps x 4 c-chunks), manual double buffer, rolled loop.
  v16bf a0 = load_a_frag(arow0, 0, kb), a1 = load_a_frag(arow1, 0, kb);
  v16bf b0 = load_bt_frag(conv_src(yb, hh, ww0, 0), 0, hi);
  v16bf b1 = load_bt_frag(conv_src(yb, hh, ww1, 0), 0, hi);
#pragma unroll 1
  for (int st = 0; st + 2 < 36; st += 2) {
    int s1 = st + 1, tap1 = s1 >> 2, c1 = (s1 & 3) * 32;
    v16bf a0n = load_a_frag(arow0, tap1 * 128 + c1, kb);
    v16bf a1n = load_a_frag(arow1, tap1 * 128 + c1, kb);
    v16bf b0n = load_bt_frag(conv_src(yb, hh, ww0, tap1), c1, hi);
    v16bf b1n = load_bt_frag(conv_src(yb, hh, ww1, tap1), c1, hi);
    acc[0][0] = wmma_bf16(a0, b0, acc[0][0]);
    acc[0][1] = wmma_bf16(a0, b1, acc[0][1]);
    acc[1][0] = wmma_bf16(a1, b0, acc[1][0]);
    acc[1][1] = wmma_bf16(a1, b1, acc[1][1]);
    int s2 = st + 2, tap2 = s2 >> 2, c2 = (s2 & 3) * 32;
    a0 = load_a_frag(arow0, tap2 * 128 + c2, kb);
    a1 = load_a_frag(arow1, tap2 * 128 + c2, kb);
    b0 = load_bt_frag(conv_src(yb, hh, ww0, tap2), c2, hi);
    b1 = load_bt_frag(conv_src(yb, hh, ww1, tap2), c2, hi);
    acc[0][0] = wmma_bf16(a0n, b0n, acc[0][0]);
    acc[0][1] = wmma_bf16(a0n, b1n, acc[0][1]);
    acc[1][0] = wmma_bf16(a1n, b0n, acc[1][0]);
    acc[1][1] = wmma_bf16(a1n, b1n, acc[1][1]);
  }
  {
    // epilogue: set A holds chunk 34 (tap 8, c 64); finish 34 and 35.
    v16bf a0n = load_a_frag(arow0, 8 * 128 + 96, kb);
    v16bf a1n = load_a_frag(arow1, 8 * 128 + 96, kb);
    v16bf b0n = load_bt_frag(conv_src(yb, hh, ww0, 8), 96, hi);
    v16bf b1n = load_bt_frag(conv_src(yb, hh, ww1, 8), 96, hi);
    acc[0][0] = wmma_bf16(a0, b0, acc[0][0]);
    acc[0][1] = wmma_bf16(a0, b1, acc[0][1]);
    acc[1][0] = wmma_bf16(a1, b0, acc[1][0]);
    acc[1][1] = wmma_bf16(a1, b1, acc[1][1]);
    acc[0][0] = wmma_bf16(a0n, b0n, acc[0][0]);
    acc[0][1] = wmma_bf16(a0n, b1n, acc[0][1]);
    acc[1][0] = wmma_bf16(a1n, b0n, acc[1][0]);
    acc[1][1] = wmma_bf16(a1n, b1n, acc[1][1]);
  }

#pragma unroll
  for (int rh = 0; rh < 2; ++rh)
#pragma unroll
    for (int chh = 0; chh < 2; ++chh)
#pragma unroll
      for (int r = 0; r < 8; ++r) {
        int o = o0 + rh * 16 + hi * 8 + r;
        float val = acc[rh][chh][r] + bo[o];
        val = val > 0.0f ? val : 0.2f * val;
        out[((size_t)(b * 128 + o)) * 65536 + hh * 256 + w0 + chh * 16 + n] = val;
      }
}

// ---------------- launch ----------------------------------------------------
extern "C" void kernel_launch(void* const* d_in, const int* in_sizes, int n_in,
                              void* d_out, int out_size, void* d_ws, size_t ws_size,
                              hipStream_t stream) {
  const float* x  = (const float*)d_in[0];
  // d_in[1] = mask (no effect per reference note)
  const float* wq = (const float*)d_in[2];
  const float* bq = (const float*)d_in[3];
  const float* wk = (const float*)d_in[4];
  const float* bk = (const float*)d_in[5];
  const float* wv = (const float*)d_in[6];
  const float* bv = (const float*)d_in[7];
  const float* wo = (const float*)d_in[8];
  const float* bo = (const float*)d_in[9];
  float* out = (float*)d_out;

  char* ws = (char*)d_ws;
  const size_t SZ_R0 = 33554432;                 // xT / y (bf16, 16.7M elems)
  __bf16* region0 = (__bf16*)ws;
  __bf16* wbf     = (__bf16*)(ws + SZ_R0);
  __bf16* wobf    = (__bf16*)(ws + SZ_R0 + 98304);
  size_t off = SZ_R0 + 98304 + 294912;
  __bf16* qb = (__bf16*)(ws + off); off += 33554432;
  __bf16* kb = (__bf16*)(ws + off); off += 33554432;
  __bf16* vT = (__bf16*)(ws + off); off += 33554432;
  float*  S  = (float *)(ws + off); off += 33554432;
  __bf16* P  = (__bf16*)(ws + off); off += 16777216;  // ~185 MB total

  k_convert_x<<<dim3(1024), dim3(256), 0, stream>>>(x, region0);
  k_convert_w<<<dim3(768), dim3(256), 0, stream>>>(wq, wk, wv, wo, wbf, wobf);
  k_qkv<<<dim3(4096, 3), dim3(128), 0, stream>>>(region0, wbf, bq, bk, bv, qb, kb, vT);
  k_scores<<<dim3(32, 8, 8), dim3(128), 0, stream>>>(qb, kb, S);
  k_softmax<<<dim3(8192), dim3(256), 0, stream>>>(S, P);
  k_pv<<<dim3(32, 16, 8), dim3(128), 0, stream>>>(P, vT, region0); // y over xT
  k_conv<<<dim3(2048, 1, 2), dim3(128), 0, stream>>>(region0, wobf, bo, out);
}
